// Attention_85710367359111
// MI455X (gfx1250) — matrix operand order repaired and checked by reading, not compiled, not run on hardware
//
#include <hip/hip_runtime.h>

// ---------------------------------------------------------------------------
// CDNA5 (gfx1250) fused attention: bf16 WMMA for all four matmuls.
// wave32, 256 threads/block = 8 waves. Software-pipelined fragment loads so
// WMMAs overlap VMEM instead of stalling on s_wait_loadcnt 0.
// ---------------------------------------------------------------------------

typedef __attribute__((ext_vector_type(16))) __bf16 v16bf;
typedef __attribute__((ext_vector_type(8)))  __bf16 v8bf;
typedef __attribute__((ext_vector_type(8)))  float  v8f;

union BF16Frag { v16bf v; v8bf h[2]; };

__device__ __forceinline__ __bf16 f2bf(float f) {
  unsigned u = __builtin_bit_cast(unsigned, f);
  u += 0x7FFFu + ((u >> 16) & 1u);              // round-to-nearest-even
  unsigned short s = (unsigned short)(u >> 16);
  return __builtin_bit_cast(__bf16, s);
}

__device__ __forceinline__ v8f wmma_bf16(v16bf a, v16bf b, v8f c) {
  // D = A(16x32 bf16) * B(32x16 bf16) + C(16x16 f32)
  return __builtin_amdgcn_wmma_f32_16x16x32_bf16(false, a, false, b,
                                                 (short)0, c, false, false);
}

// ---------------------------------------------------------------------------
// Kernel 1: f32 -> bf16 pack (grid-stride)
// ---------------------------------------------------------------------------
__global__ void __launch_bounds__(256)
cvt_f32_bf16(const float* __restrict__ src, __bf16* __restrict__ dst, int n) {
  int i = blockIdx.x * blockDim.x + threadIdx.x;
  int stride = gridDim.x * blockDim.x;
  for (; i < n; i += stride) dst[i] = f2bf(src[i]);
}

// ---------------------------------------------------------------------------
// Kernel 2: GEMM  C[M,N](f32) = A[M,K](bf16,row-major) * Bn[N,K]^T
// One wave computes a 32x64 C tile: 2 M-frags x 4 N-frags = 8 accumulators.
// K step = 32, double-buffered fragments (loads for k+32 issued before the
// WMMAs for k), so the per-WMMA wait is partial rather than loadcnt==0.
// A-operand lane layout: lane L -> m = L%16, K = (L/16)*16 .. +15 contiguous.
// B-operand lane layout: lane L -> n = L%16, K = (L/16)*16 .. +15 contiguous.
// ---------------------------------------------------------------------------
struct GFrags { BF16Frag a[2]; BF16Frag b[4]; };

__device__ __forceinline__ void load_gfrags(GFrags& f,
                                            const __bf16* __restrict__ A,
                                            const __bf16* __restrict__ Bn,
                                            long mbase, long nbase, int K,
                                            int k0, int nl, int half) {
#pragma unroll
  for (int mi = 0; mi < 2; ++mi) {
    const __bf16* ap = A + (mbase + mi * 16 + nl) * (long)K + k0 + half * 16;
    f.a[mi].h[0] = *(const v8bf*)ap;
    f.a[mi].h[1] = *(const v8bf*)(ap + 8);
  }
#pragma unroll
  for (int j = 0; j < 4; ++j) {
    const __bf16* bp = Bn + (nbase + j * 16 + nl) * (long)K + k0 + half * 16;
    f.b[j].h[0] = *(const v8bf*)bp;
    f.b[j].h[1] = *(const v8bf*)(bp + 8);
  }
}

__global__ void __launch_bounds__(256)
gemm_bf16_wmma(const __bf16* __restrict__ A, const __bf16* __restrict__ Bn,
               float* __restrict__ C, int M, int N, int K) {
  const int tid  = threadIdx.x;
  const int lane = tid & 31;
  const int wave = tid >> 5;
  const int half = lane >> 4;
  const int nl   = lane & 15;
  const int tiles_n = N >> 6;
  const int job = blockIdx.x * 8 + wave;
  const int mt  = job / tiles_n;
  const int nt  = job - mt * tiles_n;
  const long mbase = (long)mt * 32;
  const long nbase = (long)nt * 64;

  v8f zero = {0.f,0.f,0.f,0.f,0.f,0.f,0.f,0.f};
  v8f acc[2][4] = {{zero, zero, zero, zero}, {zero, zero, zero, zero}};

  GFrags cur, nxt;
  load_gfrags(cur, A, Bn, mbase, nbase, K, 0, nl, half);

#pragma unroll 2
  for (int k0 = 32; k0 < K; k0 += 32) {
    load_gfrags(nxt, A, Bn, mbase, nbase, K, k0, nl, half);
    // pull the K-step three past the one loaded here toward L2/L0 (global_prefetch_b8)
    __builtin_prefetch(A + (mbase + nl) * (long)K + k0 + 96, 0, 1);
    __builtin_prefetch(Bn + (nbase + nl) * (long)K + k0 + 96, 0, 1);
#pragma unroll
    for (int mi = 0; mi < 2; ++mi)
#pragma unroll
      for (int j = 0; j < 4; ++j)
        acc[mi][j] = wmma_bf16(cur.a[mi].v, cur.b[j].v, acc[mi][j]);
    cur = nxt;                       // SSA rename after unroll-2, no real moves
  }
#pragma unroll
  for (int mi = 0; mi < 2; ++mi)
#pragma unroll
    for (int j = 0; j < 4; ++j)
      acc[mi][j] = wmma_bf16(cur.a[mi].v, cur.b[j].v, acc[mi][j]);

#pragma unroll
  for (int mi = 0; mi < 2; ++mi)
#pragma unroll
    for (int j = 0; j < 4; ++j)
#pragma unroll
      for (int r = 0; r < 8; ++r)
        C[(mbase + mi * 16 + r + 8 * half) * (long)N + nbase + j * 16 + nl] =
            acc[mi][j][r];
}

// ---------------------------------------------------------------------------
// Kernel 3: per-(b,h,s) RMS-norm + RoPE on q,k; transpose+pack v.
// qkv layout: [B*S, 3*D], head h occupies cols h*192 + {q:0..63,k:64..127,v:128..191}
// Outputs: qb,kb bf16 [B,H,S,E]  (q pre-scaled by 1/sqrt(E))
//          vb    bf16 [B,H,E,S]  (B-operand-ready for attn*V)
// ---------------------------------------------------------------------------
__global__ void __launch_bounds__(256)
rmsrope_kernel(const float* __restrict__ qkv, const float* __restrict__ pos,
               const float* __restrict__ normw,
               __bf16* __restrict__ qb, __bf16* __restrict__ kb,
               __bf16* __restrict__ vb) {
  const int tid  = threadIdx.x;
  const int lane = tid & 31;
  const int job  = blockIdx.x * 8 + (tid >> 5);   // (b*16 + h)*2048 + s
  const int s    = job & 2047;
  const int bh   = job >> 11;
  const int b    = bh >> 4;
  const int h    = bh & 15;

  const float* base = qkv + ((size_t)(b * 2048 + s)) * 3072 + h * 192;
  const float cl = pos[s * 64 + lane];
  const float ch = pos[s * 64 + lane + 32];
  const float sl = pos[131072 + s * 64 + lane];
  const float sh = pos[131072 + s * 64 + lane + 32];
  const float w0 = normw[2 * lane];
  const float w1 = normw[2 * lane + 1];

  auto do_qk = [&](const float* t, __bf16* out, float scale) {
    float t0 = t[2 * lane], t1 = t[2 * lane + 1];
    float ss = t0 * t0 + t1 * t1;
#pragma unroll
    for (int m = 1; m < 32; m <<= 1) ss += __shfl_xor(ss, m, 32);
    float sf = rsqrtf(ss * (1.0f / 64.0f) + 1.1920929e-7f);
    float tn0 = t0 * sf * w0;                 // normalized value at index 2l
    float tn1 = t1 * sf * w1;                 // normalized value at index 2l+1
    int src0 = lane >> 1, src1 = (lane >> 1) + 16;
    float g0 = __shfl(tn0, src0, 32), g1 = __shfl(tn1, src0, 32);
    float q0 = __shfl(tn0, src1, 32), q1 = __shfl(tn1, src1, 32);
    float t_lo = (lane & 1) ? g1 : g0;        // tn[lane]
    float t_hi = (lane & 1) ? q1 : q0;        // tn[lane+32]
    float o_lo = t_lo * cl - tn1 * sl;        // y[l]    = -tn[2l+1]
    float o_hi = t_hi * ch + tn0 * sh;        // y[l+32] =  tn[2l]
    out[lane]      = f2bf(o_lo * scale);
    out[lane + 32] = f2bf(o_hi * scale);
  };

  __bf16* qo = qb + ((size_t)bh * 2048 + s) * 64;
  __bf16* ko = kb + ((size_t)bh * 2048 + s) * 64;
  do_qk(base,      qo, 0.125f);               // fold 1/sqrt(64) into q
  do_qk(base + 64, ko, 1.0f);

  const float* vt = base + 128;
  vb[((size_t)bh * 64 + lane)      * 2048 + s] = f2bf(vt[lane]);
  vb[((size_t)bh * 64 + lane + 32) * 2048 + s] = f2bf(vt[lane + 32]);
}

// ---------------------------------------------------------------------------
// Kernel 4: flash attention. One wave per (b,h, 16-query tile).
// Per 32-wide KV chunk: 4 score WMMAs + online softmax + 4 output WMMAs.
// k-fragments are double-buffered across chunks; v loads are issued before
// the softmax math so VALU work hides their latency; probs cross from
// C-layout to A-layout through a wave-private LDS tile.
// ---------------------------------------------------------------------------
__device__ __forceinline__ void load_kfrags(BF16Frag (&kf)[4],
                                            const __bf16* __restrict__ kp,
                                            int kv, int nl, int half) {
#pragma unroll
  for (int j = 0; j < 2; ++j)
#pragma unroll
    for (int ks = 0; ks < 2; ++ks) {
      const __bf16* ka = kp + (size_t)(kv + j * 16 + nl) * 64 + ks * 32 + half * 16;
      kf[j * 2 + ks].h[0] = *(const v8bf*)ka;
      kf[j * 2 + ks].h[1] = *(const v8bf*)(ka + 8);
    }
}

__global__ void __launch_bounds__(256)
attn_kernel(const __bf16* __restrict__ qb, const __bf16* __restrict__ kb,
            const __bf16* __restrict__ vb, __bf16* __restrict__ ob) {
  __shared__ __bf16 ptile[8 * 512];           // 8 waves x (16x32 bf16) = 8 KB
  const int tid  = threadIdx.x;
  const int lane = tid & 31;
  const int wave = tid >> 5;
  const int half = lane >> 4;
  const int nl   = lane & 15;
  const int job  = blockIdx.x * 8 + wave;
  const int qt   = job & 127;
  const int bh   = job >> 7;
  const int h    = bh & 15;
  const int b    = bh >> 4;
  const int q0   = qt * 16;

  const __bf16* qp = qb + ((size_t)bh * 2048 + q0) * 64;
  const __bf16* kp = kb + (size_t)bh * 2048 * 64;
  const __bf16* vp = vb + (size_t)bh * 64 * 2048;
  __bf16* pt = ptile + wave * 512;

  // q A-fragments (16x32 each), loaded once: K-step ks covers e = ks*32..+31
  BF16Frag aq[2];
#pragma unroll
  for (int ks = 0; ks < 2; ++ks) {
    const __bf16* ap = qp + nl * 64 + ks * 32 + half * 16;
    aq[ks].h[0] = *(const v8bf*)ap;
    aq[ks].h[1] = *(const v8bf*)(ap + 8);
  }

  v8f zero = {0.f,0.f,0.f,0.f,0.f,0.f,0.f,0.f};
  v8f oacc[4] = {zero, zero, zero, zero};
  float mrow[8], lrow[8];
#pragma unroll
  for (int r = 0; r < 8; ++r) { mrow[r] = -3.4e38f; lrow[r] = 0.f; }

  BF16Frag kf[4];
  load_kfrags(kf, kp, 0, nl, half);

  for (int kv = 0; kv < 2048; kv += 32) {
    // scores for 32 kv columns (two 16x16 fragments) from prefetched kf
    v8f sfr[2];
#pragma unroll
    for (int j = 0; j < 2; ++j) {
      v8f sacc = zero;
#pragma unroll
      for (int ks = 0; ks < 2; ++ks)
        sacc = wmma_bf16(aq[ks].v, kf[j * 2 + ks].v, sacc);
      sfr[j] = sacc;
    }
    // issue v loads for this chunk now; softmax VALU below hides their latency
    BF16Frag vf[4];
#pragma unroll
    for (int i = 0; i < 4; ++i) {
      const __bf16* va = vp + (size_t)(i * 16 + nl) * 2048 + kv + half * 16;
      vf[i].h[0] = *(const v8bf*)va;
      vf[i].h[1] = *(const v8bf*)(va + 8);
    }
    __builtin_prefetch(kp + (size_t)(kv + 32 + nl) * 64, 0, 1);
    __builtin_prefetch(vp + (size_t)nl * 2048 + kv + 64, 0, 1);

    // online softmax: C-frag row = r + 8*half, cols across the 16-lane group
#pragma unroll
    for (int r = 0; r < 8; ++r) {
      float cm = fmaxf(sfr[0][r], sfr[1][r]);
#pragma unroll
      for (int m = 1; m < 16; m <<= 1) cm = fmaxf(cm, __shfl_xor(cm, m, 32));
      float mn    = fmaxf(mrow[r], cm);
      float alpha = __expf(mrow[r] - mn);
      float p0 = __expf(sfr[0][r] - mn);
      float p1 = __expf(sfr[1][r] - mn);
      pt[(r + 8 * half) * 32 + nl]      = f2bf(p0);
      pt[(r + 8 * half) * 32 + 16 + nl] = f2bf(p1);
      float rs = p0 + p1;
#pragma unroll
      for (int m = 1; m < 16; m <<= 1) rs += __shfl_xor(rs, m, 32);
      lrow[r] = lrow[r] * alpha + rs;
      mrow[r] = mn;
#pragma unroll
      for (int i = 0; i < 4; ++i) oacc[i][r] *= alpha;
    }
    // reload probs in A-operand layout (wave-private LDS, DS ops are in-order)
    BF16Frag ap;
    ap.h[0] = *(const v8bf*)(pt + nl * 32 + half * 16);
    ap.h[1] = *(const v8bf*)(pt + nl * 32 + half * 16 + 8);

    // start next chunk's k loads before the O WMMAs (double buffer)
    if (kv + 32 < 2048) load_kfrags(kf, kp, kv + 32, nl, half);

    // O += P(16x32) * V(32x16) for each 16-wide slice of E
#pragma unroll
    for (int i = 0; i < 4; ++i)
      oacc[i] = wmma_bf16(ap.v, vf[i].v, oacc[i]);
  }
  // normalize and store to [B*S, D] bf16 (A operand of the output projection)
#pragma unroll
  for (int i = 0; i < 4; ++i)
#pragma unroll
    for (int r = 0; r < 8; ++r) {
      float val = oacc[i][r] / lrow[r];
      ob[((size_t)(b * 2048 + q0 + r + 8 * half)) * 1024 + h * 64 + i * 16 + nl] =
          f2bf(val);
    }
}

// ---------------------------------------------------------------------------
// Launch: pack -> QKV GEMM -> RMS/RoPE -> flash attention -> out GEMM
// Workspace layout (96 MB):
//   [0,8M)   xb   bf16 4096x1024        [8M,14M)  wqb bf16 3072x1024
//   [14M,16M) wob bf16 1024x1024        [16M,64M) qkv f32 4096x3072
//   [64M..)  qb/kb/vb bf16 8MB each     [88M,96M) obuf bf16 4096x1024
// ---------------------------------------------------------------------------
extern "C" void kernel_launch(void* const* d_in, const int* in_sizes, int n_in,
                              void* d_out, int out_size, void* d_ws, size_t ws_size,
                              hipStream_t stream) {
  (void)in_sizes; (void)n_in; (void)out_size; (void)ws_size;
  const float* x    = (const float*)d_in[0];
  const float* pos  = (const float*)d_in[1];
  const float* wqkv = (const float*)d_in[2];
  const float* wout = (const float*)d_in[3];
  const float* nw   = (const float*)d_in[4];
  float* out = (float*)d_out;

  char* w = (char*)d_ws;
  const size_t MB = 1024ull * 1024ull;
  __bf16* xb   = (__bf16*)(w + 0 * MB);
  __bf16* wqb  = (__bf16*)(w + 8 * MB);
  __bf16* wob  = (__bf16*)(w + 14 * MB);
  float*  qkvf = (float*)(w + 16 * MB);
  __bf16* qb   = (__bf16*)(w + 64 * MB);
  __bf16* kb   = (__bf16*)(w + 72 * MB);
  __bf16* vb   = (__bf16*)(w + 80 * MB);
  __bf16* obuf = (__bf16*)(w + 88 * MB);

  cvt_f32_bf16<<<2048, 256, 0, stream>>>(x,    xb,  2 * 2048 * 1024);
  cvt_f32_bf16<<<2048, 256, 0, stream>>>(wqkv, wqb, 3072 * 1024);
  cvt_f32_bf16<<<1024, 256, 0, stream>>>(wout, wob, 1024 * 1024);
  // qkv[bs,f] = x[bs,d] * w_qkv[f,d]^T : M=4096 N=3072 K=1024
  //   (4096/32)*(3072/64) = 6144 wave-tiles -> 768 blocks
  gemm_bf16_wmma<<<768, 256, 0, stream>>>(xb, wqb, qkvf, 4096, 3072, 1024);
  // 65536 rows, 8 waves/block
  rmsrope_kernel<<<8192, 256, 0, stream>>>(qkvf, pos, nw, qb, kb, vb);
  // B*H*(S/16) = 4096 wave-jobs
  attn_kernel<<<512, 256, 0, stream>>>(qb, kb, vb, obuf);
  // out[bs,o] = o[bs,d] * w_out[o,d]^T : M=4096 N=1024 K=1024
  //   (4096/32)*(1024/64) = 2048 wave-tiles -> 256 blocks
  gemm_bf16_wmma<<<256, 256, 0, stream>>>(obuf, wob, out, 4096, 1024, 1024);
}
